// NTXentLoss_54185307406888
// MI455X (gfx1250) — matrix operand order repaired and checked by reading, not compiled, not run on hardware
//
#include <hip/hip_runtime.h>
#include <hip/hip_bf16.h>
#include <math.h>

// ---- problem constants (match reference) ----
#define NROWS    4096     // batch per view
#define DDIM     256      // embed dim
#define INV_T    10.0f    // 1 / TEMPERATURE
#define TWO_N    8192.0f
#define MAXLOGIT 10.0f    // |cosine| <= 1  =>  logits <= 1/T = 10 (a-priori max)
#define NTILES   (NROWS / 16)
#define TILE_H   (16 * DDIM)          // halves per 16-row B tile (8 KB)

typedef __attribute__((ext_vector_type(16))) _Float16 v16h;
typedef __attribute__((ext_vector_type(8)))  _Float16 v8h;
typedef __attribute__((ext_vector_type(8)))  float    v8f;
typedef __attribute__((ext_vector_type(4)))  float    v4f;

// ---------------------------------------------------------------------------
// Kernel 1: L2-normalize both views to fp16, compute pos[i] = (zj_i . zi_i)/T
// One wave (32 lanes) per row; each lane owns 8 contiguous floats.
// ---------------------------------------------------------------------------
__device__ __forceinline__ float wave_allreduce_sum(float x) {
    #pragma unroll
    for (int off = 16; off > 0; off >>= 1)
        x += __shfl_xor(x, off);
    return x;
}

__global__ __launch_bounds__(256)
void ntxent_normalize_kernel(const float* __restrict__ z_i,
                             const float* __restrict__ z_j,
                             _Float16* __restrict__ v0,   // zj_hat (fp16) [N,D]
                             _Float16* __restrict__ v1,   // zi_hat (fp16) [N,D]
                             float* __restrict__ pos)     // [N], scaled by 1/T
{
    const int wave = threadIdx.x >> 5;
    const int lane = threadIdx.x & 31;
    const int row  = blockIdx.x * 8 + wave;   // grid.x = NROWS/8

    const float* zjr = z_j + (size_t)row * DDIM + lane * 8;
    const float* zir = z_i + (size_t)row * DDIM + lane * 8;

    v4f j0 = *(const v4f*)(zjr);
    v4f j1 = *(const v4f*)(zjr + 4);
    v4f i0 = *(const v4f*)(zir);
    v4f i1 = *(const v4f*)(zir + 4);

    float ssj = 0.f, ssi = 0.f;
    #pragma unroll
    for (int e = 0; e < 4; ++e) {
        ssj += j0[e] * j0[e] + j1[e] * j1[e];
        ssi += i0[e] * i0[e] + i1[e] * i1[e];
    }
    ssj = wave_allreduce_sum(ssj);
    ssi = wave_allreduce_sum(ssi);
    const float rj = rsqrtf(ssj);
    const float ri = rsqrtf(ssi);

    v8h hj, hi;
    float dotp = 0.f;
    #pragma unroll
    for (int e = 0; e < 4; ++e) {
        float fj0 = j0[e] * rj, fj1 = j1[e] * rj;
        float fi0 = i0[e] * ri, fi1 = i1[e] * ri;
        hj[e]     = (_Float16)fj0;  hj[e + 4] = (_Float16)fj1;
        hi[e]     = (_Float16)fi0;  hi[e + 4] = (_Float16)fi1;
        dotp += fj0 * fi0 + fj1 * fi1;
    }
    *(v8h*)(v0 + (size_t)row * DDIM + lane * 8) = hj;
    *(v8h*)(v1 + (size_t)row * DDIM + lane * 8) = hi;

    dotp = wave_allreduce_sum(dotp);
    if (lane == 0) pos[row] = dotp * INV_T;
}

// ---------------------------------------------------------------------------
// Kernel 2: per-view Gram matrix via WMMA, fused fixed-max logsumexp.
// Block = 128 threads = 4 waves, M=64 rows/block. The 8 KB B tile is staged
// once per block into LDS (double-buffered, fragment-order layout, one
// barrier per tile) and shared by all 4 waves -> 4x less L2 traffic.
//
// Pipeline (2x unrolled, stA/stB register ping-pong):
//   iter ct : load regs for tile ct+2 | preload 8 B frags (1 dscnt wait) |
//             8 back-to-back WMMAs | exp epilogue | store tile ct+1 regs
//             (loaded a full iteration ago) to LDS | barrier
//
// LDS layout (per buffer, halves): addr(ks, h, lane) = ks*512 + h*256 + lane*8
//   content: K = 32*ks + 16*(lane/16) + 8*h + e  of row (colBase + lane%16)
// -> every ds_load_b128 sweeps 512 contiguous bytes across all 32 lanes:
//    conflict-free on the 64 LDS banks.
// ---------------------------------------------------------------------------
__device__ __forceinline__ void gram_step(_Float16* smem, int curOff, int nxtOff,
                                          const v16h afrag[8], const v16h stage[2],
                                          const int l_dst[2], int colBase,
                                          int rowBase, int lane, int g, int ln,
                                          float s[8])
{
    // preload ALL 8 B fragments -> one dscnt wait, then back-to-back WMMAs
    v16h bf[8];
    #pragma unroll
    for (int ks = 0; ks < 8; ++ks) {
        union { v16h v; v8h h[2]; } u;
        u.h[0] = *(const v8h*)(smem + curOff + ks * 512 + lane * 8);
        u.h[1] = *(const v8h*)(smem + curOff + ks * 512 + 256 + lane * 8);
        bf[ks] = u.v;
    }

    v8f c = {};
    #pragma unroll
    for (int ks = 0; ks < 8; ++ks)
        c = __builtin_amdgcn_wmma_f32_16x16x32_f16(
                /*neg_a=*/false, afrag[ks],
                /*neg_b=*/false, bf[ks],
                /*c_mod=*/(short)0, c,
                /*reuse_a=*/false, /*reuse_b=*/false);

    // branchless fixed-max logsumexp accumulation
    const int col = colBase + ln;
    #pragma unroll
    for (int p = 0; p < 8; ++p) {
        const int row = rowBase + p + 8 * g;      // D layout: VGPR p <-> row p+8g
        float vv = fmaf(c[p], INV_T, -MAXLOGIT);  // logit - fixed max, <= 0
        vv = (col != row) ? vv : -1e30f;          // diagonal mask -> exp == 0
        s[p] += __expf(vv);
    }

    // deposit the tile staged one iteration ago into the other buffer
    #pragma unroll
    for (int r = 0; r < 2; ++r) {
        union { v16h v; v8h h[2]; } u;
        u.v = stage[r];
        *(v8h*)(smem + nxtOff + l_dst[r])       = u.h[0];
        *(v8h*)(smem + nxtOff + l_dst[r] + 256) = u.h[1];
    }
    __syncthreads();
}

__global__ __launch_bounds__(128)
void ntxent_gram_lse_kernel(const _Float16* __restrict__ v0,
                            const _Float16* __restrict__ v1,
                            const float* __restrict__ pos,
                            float* __restrict__ out)
{
    __shared__ __align__(32) _Float16 smem[2 * TILE_H];   // 2 x 8 KB B tiles

    const int tid  = threadIdx.x;
    const int wave = tid >> 5;
    const int lane = tid & 31;
    const int g    = lane >> 4;        // half-wave group
    const int ln   = lane & 15;        // lane within group

    const int rowBase = (blockIdx.x * 4 + wave) * 16;     // grid.x = NROWS/64
    const _Float16* __restrict__ R = (blockIdx.y == 0) ? v0 : v1;

    // Staging decomposition: 512 x 16B chunks per tile, 128 threads x 2 reps.
    // Virtual id v = tid + 128*r  ->  ks = v>>5, sg = (v>>4)&1, sln = v&15.
    // Global source (halves, within tile): sln*DDIM + ks*32 + sg*16  (32 B).
    // LDS dest (halves): ks*512 + h*256 + (sg*16+sln)*8   for h = 0,1.
    int g_src[2], l_dst[2];
    #pragma unroll
    for (int r = 0; r < 2; ++r) {
        const int v   = tid + 128 * r;
        const int sks = v >> 5;
        const int sg  = (v >> 4) & 1;
        const int sln = v & 15;
        g_src[r] = sln * DDIM + sks * 32 + sg * 16;
        l_dst[r] = sks * 512 + (sg * 16 + sln) * 8;
    }

    // ---- preload A fragments (16x32 fp16 per k-step), 8 k-steps of K=32 ----
    // Lane ln holds row M=ln; group g holds K in {16g..16g+15}.
    v16h afrag[8];
    {
        const _Float16* arow = R + (size_t)(rowBase + ln) * DDIM;
        #pragma unroll
        for (int ks = 0; ks < 8; ++ks) {
            union { v16h v; v8h h[2]; } u;
            u.h[0] = *(const v8h*)(arow + ks * 32 + 16 * g);
            u.h[1] = *(const v8h*)(arow + ks * 32 + 16 * g + 8);
            afrag[ks] = u.v;
        }
    }

    // ---- prologue: stage tile 0 directly into buffer 0, prefetch tile 1 ----
    #pragma unroll
    for (int r = 0; r < 2; ++r) {
        union { v16h v; v8h h[2]; } u;
        u.v = *(const v16h*)(R + g_src[r]);               // tile 0: colBase = 0
        *(v8h*)(smem + l_dst[r])       = u.h[0];
        *(v8h*)(smem + l_dst[r] + 256) = u.h[1];
    }
    v16h stA[2];
    #pragma unroll
    for (int r = 0; r < 2; ++r)                           // tile 1 into regs
        stA[r] = *(const v16h*)(R + TILE_H + g_src[r]);
    __syncthreads();

    float s[8];
    #pragma unroll
    for (int p = 0; p < 8; ++p) s[p] = 0.f;

    // ---- 2x-unrolled, one-iteration-deep staged pipeline ----
    for (int ct = 0; ct < NTILES; ct += 2) {
        v16h stB[2];
        #pragma unroll
        for (int r = 0; r < 2; ++r)                       // tile ct+2 into regs
            stB[r] = *(const v16h*)(R +
                        (size_t)(((ct + 2) & (NTILES - 1)) * 16) * DDIM + g_src[r]);
        // read buf0, deposit tile ct+1 (stA) into buf1
        gram_step((_Float16*)smem, 0, TILE_H, afrag, stA, l_dst,
                  ct * 16, rowBase, lane, g, ln, s);

        #pragma unroll
        for (int r = 0; r < 2; ++r)                       // tile ct+3 into regs
            stA[r] = *(const v16h*)(R +
                        (size_t)(((ct + 3) & (NTILES - 1)) * 16) * DDIM + g_src[r]);
        // read buf1, deposit tile ct+2 (stB) into buf0
        gram_step((_Float16*)smem, TILE_H, 0, afrag, stB, l_dst,
                  (ct + 1) * 16, rowBase, lane, g, ln, s);
    }

    // ---- sum partial exp-sums across the 16 lanes that share each row ----
    #pragma unroll
    for (int off = 8; off > 0; off >>= 1) {
        #pragma unroll
        for (int p = 0; p < 8; ++p)
            s[p] += __shfl_xor(s[p], off);
    }

    // ---- fold in the positive, emit (lse - pos) / 2n ----
    if (ln == 0) {
        float local = 0.f;
        #pragma unroll
        for (int p = 0; p < 8; ++p) {
            const int r    = rowBase + p + 8 * g;
            const float pv = pos[r];                        // already * 1/T
            const float ss = s[p] + __expf(pv - MAXLOGIT);  // add positive term
            local += (MAXLOGIT + __logf(ss)) - pv;          // lse - pos
        }
        atomicAdd(out, local * (1.0f / TWO_N));
    }
}

// ---------------------------------------------------------------------------
extern "C" void kernel_launch(void* const* d_in, const int* in_sizes, int n_in,
                              void* d_out, int out_size, void* d_ws, size_t ws_size,
                              hipStream_t stream) {
    const float* z_i = (const float*)d_in[0];
    const float* z_j = (const float*)d_in[1];
    float* out = (float*)d_out;

    _Float16* v0 = (_Float16*)d_ws;                 // zj_hat fp16  [N,D]
    _Float16* v1 = v0 + (size_t)NROWS * DDIM;       // zi_hat fp16  [N,D]
    float*   pos = (float*)(v1 + (size_t)NROWS * DDIM);  // [N]

    hipMemsetAsync(d_out, 0, sizeof(float), stream);

    ntxent_normalize_kernel<<<NROWS / 8, 256, 0, stream>>>(z_i, z_j, v0, v1, pos);

    dim3 grid(NROWS / 64, 2);   // 64 x 2 views; 4 waves/block, M=64/block
    ntxent_gram_lse_kernel<<<grid, 128, 0, stream>>>(v0, v1, pos, out);
}
